// MLDecoder_52578989638033
// MI455X (gfx1250) — matrix operand order repaired and checked by reading, not compiled, not run on hardware
//
#include <hip/hip_runtime.h>

typedef unsigned short u16;
typedef unsigned int   u32;
typedef unsigned long long u64;
typedef __attribute__((ext_vector_type(16))) __bf16 v16bf;
typedef __attribute__((ext_vector_type(8)))  __bf16 v8bf;
typedef __attribute__((ext_vector_type(8)))  float  v8f;
typedef __attribute__((ext_vector_type(4)))  unsigned int u32x4;
typedef __attribute__((ext_vector_type(8)))  int i32x8;
typedef __attribute__((ext_vector_type(4)))  int i32x4;

// ---------- helpers ----------
__device__ __forceinline__ u16 f2bf(float f) {
    u32 u = __float_as_uint(f);
    u32 r = u + 0x7FFFu + ((u >> 16) & 1u);   // round-to-nearest-even
    return (u16)(r >> 16);
}
__device__ __forceinline__ float bf2f(u16 b) {
    return __uint_as_float(((u32)b) << 16);
}
__device__ __forceinline__ v16bf cat16(v8bf lo, v8bf hi) {
    return __builtin_shufflevector(lo, hi, 0,1,2,3,4,5,6,7,8,9,10,11,12,13,14,15);
}
// A-fragment (16x32 bf16): lane holds row m; elements j<8 -> K=base+j, j>=8 -> K=base+16+(j-8)
__device__ __forceinline__ v16bf ldfragA(const u16* p) {
    v8bf lo = *(const v8bf*)p;
    v8bf hi = *(const v8bf*)(p + 16);
    return cat16(lo, hi);
}
// B-fragment (32x16 bf16): lane holds col n; elements j<8 -> K=base+j, j>=8 -> K=base+16+(j-8)
__device__ __forceinline__ v16bf ldfragB(const u16* p) {
    v8bf lo = *(const v8bf*)p;
    v8bf hi = *(const v8bf*)(p + 16);
    return cat16(lo, hi);
}
__device__ __forceinline__ v8f wmma_bf16(v16bf a, v16bf b, v8f c) {
    return __builtin_amdgcn_wmma_f32_16x16x32_bf16(false, a, false, b, (short)0, c, false, false);
}
#define V8F_ZERO {0.f,0.f,0.f,0.f,0.f,0.f,0.f,0.f}

// async 16B copy global -> LDS (CDNA5 GLOBAL_LOAD_ASYNC_TO_LDS_B128, ASYNCcnt)
__device__ __forceinline__ void async_g2l_b128(u32 lds_off, const u16* gaddr) {
    asm volatile("global_load_async_to_lds_b128 %0, %1, off"
                 :: "v"(lds_off), "v"(gaddr) : "memory");
}
__device__ __forceinline__ void wait_async0() {
    asm volatile("s_wait_asynccnt 0" ::: "memory");
}

// TDM: load a 64-row x 32-bf16 2D tile (row stride = lda elements) into LDS,
// padded by the TDM itself to a 40-element LDS row stride (pad 4 DWORDs every 16).
__device__ __forceinline__ void tdm_load_tile64x32(const u16* gptr, u32 lds, int lda) {
    u64 ga = (u64)(uintptr_t)gptr;                       // byte address of tile start
    u32x4 g0 = { 1u,                                     // count=1 (valid descriptor)
                 lds,                                    // lds_addr (bytes)
                 (u32)ga,                                // global_addr[31:0]
                 (u32)((ga >> 32) & 0x1FFFFFFu) | 0x80000000u };  // addr[56:32] | type=2
    i32x8 g1 = { (int)((1u << 16)     // data_size = 2 bytes
                     | (1u << 20)     // pad_enable
                     | (3u << 22)     // pad_interval: every 16 DWORDs (64B row)
                     | (3u << 25)),   // pad_amount: 4 DWORDs (16B)
                 (int)(32u << 16),    // tensor_dim0[15:0]=32 (w1[31:16])
                 (int)(64u << 16),    // tensor_dim0[31:16]=0, tensor_dim1[15:0]=64
                 (int)(32u << 16),    // tensor_dim1[31:16]=0, tile_dim0=32
                 64,                  // tile_dim1=64, tile_dim2=0
                 lda,                 // tensor_dim0_stride[31:0] (element units)
                 0, 0 };
    i32x4 z4 = {0, 0, 0, 0};
    i32x8 z8 = {0, 0, 0, 0, 0, 0, 0, 0};
    __builtin_amdgcn_tensor_load_to_lds(g0, g1, z4, z4, z8, 0);
}

// ---------- fp32 -> bf16 convert (vectorized) ----------
struct __align__(8) us4 { u16 x, y, z, w; };
__global__ void cvt_f32_bf16_v4(const float4* __restrict__ in, us4* __restrict__ out, int n4) {
    for (int i = blockIdx.x * blockDim.x + threadIdx.x; i < n4; i += gridDim.x * blockDim.x) {
        float4 f = in[i];
        us4 o;
        o.x = f2bf(f.x); o.y = f2bf(f.y); o.z = f2bf(f.z); o.w = f2bf(f.w);
        out[i] = o;
    }
}

// ---------- t = LayerNorm(2*query) -> bf16 [256,768] ----------
__global__ __launch_bounds__(256) void ln2_query(
    const float* __restrict__ q, const float* __restrict__ gam,
    const float* __restrict__ bet, u16* __restrict__ t_bf)
{
    __shared__ float red[256];
    const int row = blockIdx.x, tid = threadIdx.x;
    const float* qr = q + (size_t)row * 768;
    float x0 = 2.f * qr[tid], x1 = 2.f * qr[tid + 256], x2 = 2.f * qr[tid + 512];
    red[tid] = x0 + x1 + x2; __syncthreads();
    for (int o = 128; o > 0; o >>= 1) { if (tid < o) red[tid] += red[tid + o]; __syncthreads(); }
    float mean = red[0] * (1.f / 768.f);
    __syncthreads();
    float d0 = x0 - mean, d1 = x1 - mean, d2 = x2 - mean;
    red[tid] = d0 * d0 + d1 * d1 + d2 * d2; __syncthreads();
    for (int o = 128; o > 0; o >>= 1) { if (tid < o) red[tid] += red[tid + o]; __syncthreads(); }
    float inv = rsqrtf(red[0] * (1.f / 768.f) + 1e-5f);
    u16* outp = t_bf + (size_t)row * 768;
    outp[tid]       = f2bf(d0 * inv * gam[tid]       + bet[tid]);
    outp[tid + 256] = f2bf(d1 * inv * gam[tid + 256] + bet[tid + 256]);
    outp[tid + 512] = f2bf(d2 * inv * gam[tid + 512] + bet[tid + 512]);
}

// ---------- out = LayerNorm(A + Bres[row & mask]) -> bf16, D=768 ----------
__global__ __launch_bounds__(256) void add_ln(
    const u16* __restrict__ A, const u16* __restrict__ Br, unsigned mask,
    const float* __restrict__ gam, const float* __restrict__ bet,
    u16* __restrict__ out)
{
    __shared__ float red[256];
    const int row = blockIdx.x, tid = threadIdx.x;
    const u16* ar = A + (size_t)row * 768;
    const u16* br = Br + (size_t)(row & mask) * 768;
    float x0 = bf2f(ar[tid])       + bf2f(br[tid]);
    float x1 = bf2f(ar[tid + 256]) + bf2f(br[tid + 256]);
    float x2 = bf2f(ar[tid + 512]) + bf2f(br[tid + 512]);
    red[tid] = x0 + x1 + x2; __syncthreads();
    for (int o = 128; o > 0; o >>= 1) { if (tid < o) red[tid] += red[tid + o]; __syncthreads(); }
    float mean = red[0] * (1.f / 768.f);
    __syncthreads();
    float d0 = x0 - mean, d1 = x1 - mean, d2 = x2 - mean;
    red[tid] = d0 * d0 + d1 * d1 + d2 * d2; __syncthreads();
    for (int o = 128; o > 0; o >>= 1) { if (tid < o) red[tid] += red[tid + o]; __syncthreads(); }
    float inv = rsqrtf(red[0] * (1.f / 768.f) + 1e-5f);
    u16* outp = out + (size_t)row * 768;
    outp[tid]       = f2bf(d0 * inv * gam[tid]       + bet[tid]);
    outp[tid + 256] = f2bf(d1 * inv * gam[tid + 256] + bet[tid + 256]);
    outp[tid + 512] = f2bf(d2 * inv * gam[tid + 512] + bet[tid + 512]);
}

// ---------- generic bf16 WMMA GEMM: C = act(A[M,K] @ B[K,N] + bias) ----------
// grid.x = M/64, grid.y = N/128, 256 threads (8 waves, each wave a 16x64 output).
// Double-buffered LDS; A tile via TDM (tensor_load_to_lds); B via uint4 load + scatter.
// Requires: M%64==0, K%32==0, N%128==0, ldb%8==0, bias != null (all call sites comply).
__global__ __launch_bounds__(256) void gemm_bf16_wmma(
    const u16* __restrict__ A, int lda,
    const u16* __restrict__ Bm, int ldb,
    const float* __restrict__ bias,
    u16* __restrict__ C, int ldc, int K, int relu)
{
    __shared__ __align__(16) u16 As[2][64 * 40];    // [m][k], stride 40 (TDM pads)
    __shared__ __align__(16) u16 Bs[2][128 * 40];   // [n][k], stride 40 (B transposed in LDS)
    const int m0 = blockIdx.x * 64, n0 = blockIdx.y * 128;
    const int tid = threadIdx.x;
    const int wave = tid >> 5, lane = tid & 31;
    const int wm = wave & 3, wn = wave >> 2;        // wn selects 64-wide N half
    const int l15 = lane & 15, hi = lane >> 4;
    const int kb = tid >> 3, ncb = (tid & 7) * 8;   // B coop load: k kb, 8 n's
    v8f acc[4] = {V8F_ZERO, V8F_ZERO, V8F_ZERO, V8F_ZERO};

    const u16* gB = Bm + (size_t)kb * ldb + n0 + ncb;   // advance by k0*ldb

    auto stageA = [&](int k0, int buf) {
        if (wave == 0)
            tdm_load_tile64x32(A + (size_t)m0 * lda + k0,
                               (u32)(uintptr_t)&As[buf][0], lda);
    };
    auto stageB = [&](int k0, int buf) {
        union { uint4 q; u16 h[8]; } u0, u1;
        u0.q = *(const uint4*)(gB + (size_t)k0 * ldb);
        u1.q = *(const uint4*)(gB + (size_t)k0 * ldb + 64);
        #pragma unroll
        for (int j = 0; j < 8; ++j) {
            Bs[buf][(ncb + j) * 40 + kb]      = u0.h[j];
            Bs[buf][(64 + ncb + j) * 40 + kb] = u1.h[j];
        }
    };

    stageA(0, 0);
    stageB(0, 0);
    __builtin_amdgcn_s_wait_tensorcnt(0);   // no-op for non-issuing waves (TENSORcnt==0)
    __syncthreads();

    int buf = 0;
    for (int k0 = 0; k0 < K; k0 += 32) {
        int nb = buf ^ 1;
        if (k0 + 32 < K) { stageA(k0 + 32, nb); stageB(k0 + 32, nb); }
        // load all fragments first, then issue the 4 WMMAs
        v16bf a  = ldfragA(&As[buf][(wm * 16 + l15) * 40 + hi * 8]);
        const u16* bbase = &Bs[buf][(wn * 64 + l15) * 40 + hi * 8];
        v16bf b0 = ldfragB(bbase);
        v16bf b1 = ldfragB(bbase + 16 * 40);
        v16bf b2 = ldfragB(bbase + 32 * 40);
        v16bf b3 = ldfragB(bbase + 48 * 40);
        acc[0] = wmma_bf16(a, b0, acc[0]);
        acc[1] = wmma_bf16(a, b1, acc[1]);
        acc[2] = wmma_bf16(a, b2, acc[2]);
        acc[3] = wmma_bf16(a, b3, acc[3]);
        __builtin_amdgcn_s_wait_tensorcnt(0);
        __syncthreads();
        buf = nb;
    }
    // epilogue: bias preloaded once per lane, fully unrolled stores
    float bv[4];
    #pragma unroll
    for (int t = 0; t < 4; ++t) bv[t] = bias[n0 + wn * 64 + t * 16 + l15];
    u16* crow = C + (size_t)(m0 + wm * 16 + 8 * hi) * ldc + (n0 + wn * 64 + l15);
    #pragma unroll
    for (int t = 0; t < 4; ++t)
        #pragma unroll
        for (int r = 0; r < 8; ++r) {
            float vv = acc[t][r] + bv[t];
            if (relu) vv = fmaxf(vv, 0.f);
            crow[(size_t)r * ldc + t * 16] = f2bf(vv);
        }
}

// ---------- fused cross-attention: one wave per (b, h, 16-query tile) ----------
// q:[256,768] bf16  k,v:[128*49,768] bf16  o:[128*256,768] bf16
__global__ __launch_bounds__(32) void attn_fused(
    const u16* __restrict__ q, const u16* __restrict__ k,
    const u16* __restrict__ v, u16* __restrict__ o)
{
    const int gt = blockIdx.x, hh = blockIdx.y, bb = blockIdx.z;
    const int tid = threadIdx.x;
    const int g0 = gt * 16;

    __shared__ __align__(16) u16 Qs[16 * 104];   // [g][d]
    __shared__ __align__(16) u16 Ks[64 * 104];   // [n][d]  (B-layout for scores = q @ k^T)
    __shared__ __align__(16) u16 Vs[96 * 72];    // [d][n]  (B-layout for attn @ v)
    __shared__ __align__(16) u16 Aat[16 * 72];   // [g][n]  attn in A-layout

    for (int i = tid; i < 16 * 96; i += 32) {
        int gl = i / 96, d = i % 96;
        Qs[gl * 104 + d] = q[(size_t)(g0 + gl) * 768 + hh * 96 + d];
    }
    for (int i = tid; i < 64 * 96; i += 32) {
        int n = i / 96, d = i % 96;
        int nc = n < 49 ? n : 48;                       // clamp address, select value
        u16 val = k[((size_t)bb * 49 + nc) * 768 + hh * 96 + d];
        Ks[n * 104 + d] = (n < 49) ? val : (u16)0;
    }
    for (int i = tid; i < 96 * 64; i += 32) {
        int d = i / 64, n = i % 64;
        int nc = n < 49 ? n : 48;
        u16 val = v[((size_t)bb * 49 + nc) * 768 + hh * 96 + d];
        Vs[d * 72 + n] = (n < 49) ? val : (u16)0;
    }
    __syncthreads();

    const int l15 = tid & 15, hi = tid >> 4;

    // scores[16 g, 64 n] = q_tile @ k^T, K = 96
    v8f sc[4];
    #pragma unroll
    for (int t = 0; t < 4; ++t) sc[t] = V8F_ZERO;
    #pragma unroll
    for (int k0 = 0; k0 < 96; k0 += 32) {
        v16bf a = ldfragA(&Qs[l15 * 104 + k0 + hi * 8]);
        #pragma unroll
        for (int t = 0; t < 4; ++t) {
            v16bf bm = ldfragB(&Ks[(t * 16 + l15) * 104 + k0 + hi * 8]);
            sc[t] = wmma_bf16(a, bm, sc[t]);
        }
    }
    // scale + mask (padded n >= 49)
    const float scale = 0.1020620726f;  // 1/sqrt(96)
    #pragma unroll
    for (int t = 0; t < 4; ++t) {
        int ng = t * 16 + l15;
        #pragma unroll
        for (int r = 0; r < 8; ++r)
            sc[t][r] = (ng < 49) ? sc[t][r] * scale : -1e30f;
    }
    // softmax over n: row m = r + 8*hi lives in one 16-lane half -> shfl_xor within half
    #pragma unroll
    for (int r = 0; r < 8; ++r) {
        float mx = fmaxf(fmaxf(sc[0][r], sc[1][r]), fmaxf(sc[2][r], sc[3][r]));
        #pragma unroll
        for (int off = 8; off >= 1; off >>= 1) mx = fmaxf(mx, __shfl_xor(mx, off, 32));
        float s = 0.f;
        #pragma unroll
        for (int t = 0; t < 4; ++t) { float e = __expf(sc[t][r] - mx); sc[t][r] = e; s += e; }
        #pragma unroll
        for (int off = 8; off >= 1; off >>= 1) s += __shfl_xor(s, off, 32);
        float inv = 1.f / s;
        #pragma unroll
        for (int t = 0; t < 4; ++t) sc[t][r] *= inv;
    }
    // transpose attn (C-layout) -> A-layout via LDS
    #pragma unroll
    for (int t = 0; t < 4; ++t)
        #pragma unroll
        for (int r = 0; r < 8; ++r)
            Aat[(r + 8 * hi) * 72 + t * 16 + l15] = f2bf(sc[t][r]);
    __syncthreads();

    // o[16 g, 96 d] = attn @ v, K = 64
    v8f oc[6];
    #pragma unroll
    for (int t = 0; t < 6; ++t) oc[t] = V8F_ZERO;
    #pragma unroll
    for (int k0 = 0; k0 < 64; k0 += 32) {
        v16bf a = ldfragA(&Aat[l15 * 72 + k0 + hi * 8]);
        #pragma unroll
        for (int t = 0; t < 6; ++t) {
            v16bf bm = ldfragB(&Vs[(t * 16 + l15) * 72 + k0 + hi * 8]);
            oc[t] = wmma_bf16(a, bm, oc[t]);
        }
    }
    #pragma unroll
    for (int t = 0; t < 6; ++t)
        #pragma unroll
        for (int r = 0; r < 8; ++r) {
            int g = g0 + r + 8 * hi;
            int d = t * 16 + l15;
            o[((size_t)bb * 256 + g) * 768 + hh * 96 + d] = f2bf(oc[t][r]);
        }
}

// ---------- GroupFC: out[b, g*50+f] = h[b,g,:] @ Wg[g,:,f] + bg ----------
// grid (mtile=2, g=256), 256 threads; A row stride = 256*768
__global__ __launch_bounds__(256) void groupfc_wmma(
    const u16* __restrict__ h, const u16* __restrict__ wg,
    const float* __restrict__ bg, float* __restrict__ out)
{
    __shared__ __align__(16) u16 As[64 * 40];
    __shared__ __align__(16) u16 Bs[64 * 40];
    const int mt = blockIdx.x, g = blockIdx.y;
    const int tid = threadIdx.x;
    const int wave = tid >> 5, lane = tid & 31;
    const int wm = wave & 3, wn = wave >> 2;
    const int l15 = lane & 15, hi = lane >> 4;
    const int ra = tid >> 2, kca = (tid & 3) * 8;
    const int kb = tid >> 3, ncb = (tid & 7) * 8;
    const u16* Abase = h + (size_t)g * 768;
    const u16* Bbase = wg + (size_t)g * 768 * 50;
    const int m0 = mt * 64;
    v8f acc0 = V8F_ZERO, acc1 = V8F_ZERO;

    for (int k0 = 0; k0 < 768; k0 += 32) {
        {   // A tile via async global->LDS DMA
            u32 lds = (u32)(uintptr_t)&As[ra * 40 + kca];
            async_g2l_b128(lds, Abase + (size_t)(m0 + ra) * (256 * 768) + k0 + kca);
        }
        #pragma unroll
        for (int j = 0; j < 8; ++j) {                 // clamped loads, no exec divergence
            int n = ncb + j;
            int nc = n < 50 ? n : 49;
            u16 val = Bbase[(size_t)(k0 + kb) * 50 + nc];
            Bs[n * 40 + kb] = (n < 50) ? val : (u16)0;
        }
        wait_async0();
        __syncthreads();
        v16bf a = ldfragA(&As[(wm * 16 + l15) * 40 + hi * 8]);
        const u16* bbase = &Bs[(wn * 32 + l15) * 40 + hi * 8];
        v16bf b0 = ldfragB(bbase);
        v16bf b1 = ldfragB(bbase + 16 * 40);
        acc0 = wmma_bf16(a, b0, acc0);
        acc1 = wmma_bf16(a, b1, acc1);
        __syncthreads();
    }
    #pragma unroll
    for (int t = 0; t < 2; ++t) {
        v8f acc = t ? acc1 : acc0;
        #pragma unroll
        for (int r = 0; r < 8; ++r) {
            int b = m0 + wm * 16 + r + 8 * hi;
            int n = wn * 32 + t * 16 + l15;
            int cls = g * 50 + n;
            if (n < 50 && cls < 12547)
                out[(size_t)b * 12547 + cls] = acc[r] + bg[cls];
        }
    }
}

// ---------- host ----------
extern "C" void kernel_launch(void* const* d_in, const int* in_sizes, int n_in,
                              void* d_out, int out_size, void* d_ws, size_t ws_size,
                              hipStream_t stream) {
    (void)in_sizes; (void)n_in; (void)out_size; (void)ws_size;
    const float* x       = (const float*)d_in[0];
    const float* W_embed = (const float*)d_in[1];
    const float* b_embed = (const float*)d_in[2];
    const float* query   = (const float*)d_in[3];
    const float* Wq = (const float*)d_in[4];  const float* bq  = (const float*)d_in[5];
    const float* Wk = (const float*)d_in[6];  const float* bk  = (const float*)d_in[7];
    const float* Wv = (const float*)d_in[8];  const float* bv  = (const float*)d_in[9];
    const float* Wo = (const float*)d_in[10]; const float* bo  = (const float*)d_in[11];
    const float* g1 = (const float*)d_in[12]; const float* be1 = (const float*)d_in[13];
    const float* g2 = (const float*)d_in[14]; const float* be2 = (const float*)d_in[15];
    const float* g3 = (const float*)d_in[16]; const float* be3 = (const float*)d_in[17];
    const float* W1 = (const float*)d_in[18]; const float* b1  = (const float*)d_in[19];
    const float* W2 = (const float*)d_in[20]; const float* b2  = (const float*)d_in[21];
    const float* Wg = (const float*)d_in[22]; const float* bg  = (const float*)d_in[23];
    float* out = (float*)d_out;

    // bump allocator over workspace (bf16 intermediates)
    char* p = (char*)d_ws;
    auto alloc = [&](size_t elems) -> u16* {
        u16* r = (u16*)p;
        p += ((elems * 2 + 255) & ~(size_t)255);
        return r;
    };
    u16* xb   = alloc(12845056);  // x bf16 [6272,2048]
    u16* wEb  = alloc(1572864);
    u16* wqb  = alloc(589824);
    u16* wkb  = alloc(589824);
    u16* wvb  = alloc(589824);
    u16* wob  = alloc(589824);
    u16* w1b  = alloc(1572864);
    u16* w2b  = alloc(1572864);
    u16* wgb  = alloc(9830400);
    u16* tb   = alloc(196608);    // LN(2*query) [256,768]
    u16* qb   = alloc(196608);    // q proj
    u16* memb = alloc(4816896);   // relu embed [6272,768]
    u16* kbuf = alloc(4816896);
    u16* vbuf = alloc(4816896);
    u16* ob   = alloc(25165824);  // attention out [32768,768]
    u16* ff1b = alloc(67108864);  // FFN hidden [32768,2048]; also aliases o-proj output
    u16* t2b  = alloc(25165824);  // norm2 out
    u16* ff2b = alloc(25165824);  // FFN out
    u16* hb   = alloc(25165824);  // norm3 out

    auto cvt = [&](const float* src, u16* dst, int n) {
        cvt_f32_bf16_v4<<<dim3(1024), dim3(256), 0, stream>>>(
            (const float4*)src, (us4*)dst, n / 4);
    };
    cvt(x, xb, 12845056);
    cvt(W_embed, wEb, 1572864);
    cvt(Wq, wqb, 589824);
    cvt(Wk, wkb, 589824);
    cvt(Wv, wvb, 589824);
    cvt(Wo, wob, 589824);
    cvt(W1, w1b, 1572864);
    cvt(W2, w2b, 1572864);
    cvt(Wg, wgb, 9830400);

    ln2_query<<<dim3(256), dim3(256), 0, stream>>>(query, g1, be1, tb);

    // mem = relu(x @ W_embed + b_embed)   [6272,768]
    gemm_bf16_wmma<<<dim3(98, 6), 256, 0, stream>>>(xb, 2048, wEb, 768, b_embed, memb, 768, 2048, 1);
    // q = t @ Wq + bq   [256,768]
    gemm_bf16_wmma<<<dim3(4, 6), 256, 0, stream>>>(tb, 768, wqb, 768, bq, qb, 768, 768, 0);
    // k, v = mem @ Wk/Wv + b   [6272,768]
    gemm_bf16_wmma<<<dim3(98, 6), 256, 0, stream>>>(memb, 768, wkb, 768, bk, kbuf, 768, 768, 0);
    gemm_bf16_wmma<<<dim3(98, 6), 256, 0, stream>>>(memb, 768, wvb, 768, bv, vbuf, 768, 768, 0);
    // fused attention -> ob [32768,768]
    attn_fused<<<dim3(16, 8, 128), dim3(32), 0, stream>>>(qb, kbuf, vbuf, ob);
    // o-proj = ob @ Wo + bo  (into ff1b region, free at this point)
    gemm_bf16_wmma<<<dim3(512, 6), 256, 0, stream>>>(ob, 768, wob, 768, bo, ff1b, 768, 768, 0);
    // t2 = LN(o-proj + t[g])   (t broadcast over batch: row & 255)
    add_ln<<<dim3(32768), dim3(256), 0, stream>>>(ff1b, tb, 255u, g2, be2, t2b);
    // ff1 = relu(t2 @ W1 + b1)  [32768,2048]
    gemm_bf16_wmma<<<dim3(512, 16), 256, 0, stream>>>(t2b, 768, w1b, 2048, b1, ff1b, 2048, 768, 1);
    // ff2 = ff1 @ W2 + b2  [32768,768]
    gemm_bf16_wmma<<<dim3(512, 6), 256, 0, stream>>>(ff1b, 2048, w2b, 768, b2, ff2b, 768, 2048, 0);
    // h = LN(ff2 + t2)
    add_ln<<<dim3(32768), dim3(256), 0, stream>>>(ff2b, t2b, 0xFFFFFFFFu, g3, be3, hb);
    // logits
    groupfc_wmma<<<dim3(2, 256), 256, 0, stream>>>(hb, wgb, bg, out);
}
